// att_27848567948133
// MI455X (gfx1250) — matrix operand order repaired and checked by reading, not compiled, not run on hardware
//
#include <hip/hip_runtime.h>

// ---------------- types ----------------
typedef __attribute__((ext_vector_type(16))) _Float16 v16h;
typedef __attribute__((ext_vector_type(4)))  _Float16 h4;
typedef __attribute__((ext_vector_type(8)))  float    v8f;
typedef __attribute__((ext_vector_type(4)))  unsigned u32x4;
typedef __attribute__((ext_vector_type(4)))  float    f32x4;

#define NTOK 2304     // H*W = 48*48
#define NB   4
#define NC   64
#define HW48 48

// ---------------- attend kernel dynamic LDS layout ----------------
#define SM_E      0                        /* energy 16 x 2304 f32 = 147456 */
#define SM_HIST   (16 * NTOK * 4)          /* 8 waves x 256-bin histograms  */
#define SM_PART   (SM_HIST + 8 * 256 * 4)  /* 8 waves x 32 lanes x 8 f32    */
#define SM_TOTAL  (SM_PART + 8192)         /* = 163840 B: 2 blocks = 320KB  */

__device__ __forceinline__ unsigned f2ord(float f) {
  unsigned u = __float_as_uint(f);
  return (u & 0x80000000u) ? ~u : (u | 0x80000000u);
}

__device__ __forceinline__ float wave_max(float v) {
#pragma unroll
  for (int m = 16; m >= 1; m >>= 1) v = fmaxf(v, __shfl_xor(v, m, 32));
  return v;
}
__device__ __forceinline__ float wave_sum(float v) {
#pragma unroll
  for (int m = 16; m >= 1; m >>= 1) v += __shfl_xor(v, m, 32);
  return v;
}
__device__ __forceinline__ unsigned wave_or(unsigned v) {
#pragma unroll
  for (int m = 16; m >= 1; m >>= 1) v |= __shfl_xor((int)v, m, 32);
  return v;
}

// qT: [B][N][64] f16 (q pre-scaled by 1/sqrt(C)); kT: [B][N][64] f16; vH: [B][64][N] f16
// out: [B][64][N] f32. One block = 16 query rows of one batch. 256 threads = 8 waves.
__global__ void __launch_bounds__(256) attend_kernel(
    const _Float16* __restrict__ qT,
    const _Float16* __restrict__ kT,
    const _Float16* __restrict__ vH,
    float* __restrict__ out,
    const int* __restrict__ topkPtr)
{
  extern __shared__ __align__(16) char smem[];
  float*    e     = (float*)(smem + SM_E);
  unsigned* histB = (unsigned*)(smem + SM_HIST);
  float*    part  = (float*)(smem + SM_PART);

  const int tid  = threadIdx.x;
  const int w    = tid >> 5;
  const int lane = tid & 31;
  const int h    = lane >> 4;   // half-wave select per WMMA layout
  const int lm   = lane & 15;
  const int b    = blockIdx.y;
  const int n0   = blockIdx.x * 16;

  union Frag { u32x4 u[2]; v16h v; };

  // ---- A fragments for the query tile (invariant over key tiles) ----
  // A 16x32 f16 layout: halves 0..7 -> K = 8h + i ; halves 8..15 -> K = 16 + 8h + (i-8)
  const u32x4* qb = (const u32x4*)(qT + ((size_t)(b * NTOK + n0 + lm) << 6));
  Frag a0, a1;
  a0.u[0] = qb[h];     a0.u[1] = qb[2 + h];   // K chunk 0..31
  a1.u[0] = qb[4 + h]; a1.u[1] = qb[6 + h];   // K chunk 32..63

  // ---- phase 2: energy tile [16 x 2304] into LDS via WMMA f16 ----
  for (int t = w; t < NTOK / 16; t += 8) {
    const int ncol = t * 16 + lm;
    // B 32x16 f16 layout: halves 0..7 -> K = 8h + i ; halves 8..15 -> K = 16 + 8h + (i-8) (+32 per chunk)
    const u32x4* kb = (const u32x4*)(kT + ((size_t)(b * NTOK + ncol) << 6));
    Frag b0, b1;
    b0.u[0] = kb[h];     b0.u[1] = kb[2 + h];
    b1.u[0] = kb[4 + h]; b1.u[1] = kb[6 + h];
    v8f acc = {0.f, 0.f, 0.f, 0.f, 0.f, 0.f, 0.f, 0.f};
    acc = __builtin_amdgcn_wmma_f32_16x16x32_f16(false, a0.v, false, b0.v, (short)0, acc, false, false);
    acc = __builtin_amdgcn_wmma_f32_16x16x32_f16(false, a1.v, false, b1.v, (short)0, acc, false, false);
#pragma unroll
    for (int r = 0; r < 8; ++r)                  // C/D: M = r + 8h, N = lm
      e[(r + 8 * h) * NTOK + ncol] = acc[r];
  }
  __syncthreads();

  // ---- phase 3+4: wave-per-row exact top-k (8-bit radix select) + masked softmax ----
  // Wave w owns rows {2w, 2w+1} and a private 256-bin histogram -> no block barriers.
  // All row sweeps use vectorized ds_load_b128 (lane owns 4 consecutive floats/step).
  int kk = topkPtr ? *topkPtr : 100;
  kk = kk < 1 ? 1 : (kk > NTOK ? NTOK : kk);
  unsigned* myh = histB + w * 256;
  for (int rr = 0; rr < 2; ++rr) {
    const int r = w * 2 + rr;
    float* erow = e + r * NTOK;
    unsigned pref = 0, rem = (unsigned)kk;
    for (int pass = 0; pass < 4; ++pass) {
      const int shift = 24 - 8 * pass;
#pragma unroll
      for (int i = 0; i < 8; ++i) myh[lane * 8 + i] = 0u;   // lane owns bins [8L, 8L+7]
      __threadfence_block();
      for (int cI = 0; cI < NTOK / 128; ++cI) {
        const f32x4 ev4 = *(const f32x4*)(erow + cI * 128 + lane * 4);
#pragma unroll
        for (int q = 0; q < 4; ++q) {
          unsigned u = f2ord(ev4[q]);
          if (pass == 0 || (u >> (shift + 8)) == pref)
            atomicAdd(&myh[(u >> shift) & 255u], 1u);
        }
      }
      __threadfence_block();
      unsigned c[8], loc = 0;
#pragma unroll
      for (int i = 0; i < 8; ++i) { c[i] = myh[lane * 8 + i]; loc += c[i]; }
      // inclusive suffix-sum over lanes (lane 31 = highest bins)
      unsigned incl = loc;
#pragma unroll
      for (int ofs = 1; ofs < 32; ofs <<= 1) {
        unsigned nb = (unsigned)__shfl_down((int)incl, ofs, 32);
        if (lane + ofs < 32) incl += nb;
      }
      const unsigned above = incl - loc;  // elements in bins strictly higher than mine
      unsigned pk = 0;
      if (above < rem && rem <= incl) {   // exactly one winner lane
        unsigned cum = above;
#pragma unroll
        for (int i = 7; i >= 0; --i) {
          cum += c[i];
          if (cum >= rem) { pk = ((unsigned)(lane * 8 + i) << 16) | (rem - (cum - c[i])); break; }
        }
      }
      pk = wave_or(pk);                   // broadcast {digit, new remaining}
      pref = (pref << 8) | (pk >> 16);
      rem = pk & 0xFFFFu;
    }
    const unsigned thrU = pref;           // exact bits of kth-largest energy
    float lmax = -3.4e38f;
    for (int cI = 0; cI < NTOK / 128; ++cI) {
      const f32x4 ev4 = *(const f32x4*)(erow + cI * 128 + lane * 4);
      lmax = fmaxf(lmax, fmaxf(fmaxf(ev4[0], ev4[1]), fmaxf(ev4[2], ev4[3])));
    }
    const float rmax = wave_max(lmax);    // global row max == masked max (k >= 1)
    float lsum = 0.f;
    for (int cI = 0; cI < NTOK / 128; ++cI) {
      const f32x4 ev4 = *(const f32x4*)(erow + cI * 128 + lane * 4);
#pragma unroll
      for (int q = 0; q < 4; ++q)
        if (f2ord(ev4[q]) >= thrU) lsum += __expf(ev4[q] - rmax);
    }
    const float inv = 1.f / wave_sum(lsum);
    // Normalize and store attn weights *as f16 in place* (first half of the row
    // region). Safe within a wave: per-iteration f32 read precedes the f16 store
    // (LDS ops are wave-wide and in-order); iteration c overwrites floats
    // [64c, 64c+63], which were fully read at iterations <= c/2.
    _Float16* erowH = (_Float16*)erow;
    for (int cI = 0; cI < NTOK / 128; ++cI) {
      const int j = cI * 128 + lane * 4;
      const f32x4 ev4 = *(const f32x4*)(erow + j);
      h4 a4;
#pragma unroll
      for (int q = 0; q < 4; ++q) {
        float ev = ev4[q];
        float a = (f2ord(ev) >= thrU) ? __expf(ev - rmax) * inv : 0.f;
        a4[q] = (_Float16)a;
      }
      *(h4*)(erowH + j) = a4;
    }
  }
  __syncthreads();

  // ---- phase 5: out[c, nq] = sum_k v[c,k] * attn[nq,k] via WMMA ----
  // wave w: channel tile (w&3), K-half (w>>2) of 2304 -> 36 chunks of 32.
  // A-fragment now reads pre-converted f16 attn straight from LDS (2x b128).
  const int c0 = (w & 3) * 16;
  const int khalf = w >> 2;
  v8f dacc = {0.f, 0.f, 0.f, 0.f, 0.f, 0.f, 0.f, 0.f};
  const _Float16* arowH = (const _Float16*)(e + lm * NTOK);  // A rows are queries (M = lm)
  for (int ch = khalf * 36; ch < khalf * 36 + 36; ++ch) {
    const int k0 = ch * 32;
    Frag af;
    af.u[0] = *(const u32x4*)(arowH + k0 + 8 * h);        // halves: K = 8h + i
    af.u[1] = *(const u32x4*)(arowH + k0 + 16 + 8 * h);   // halves: K = 16 + 8h + i
    const u32x4* vb = (const u32x4*)(vH + (size_t)(b * NC + c0 + lm) * NTOK + k0 + 8 * h);
    Frag bf; bf.u[0] = vb[0]; bf.u[1] = vb[2];
    dacc = __builtin_amdgcn_wmma_f32_16x16x32_f16(false, af.v, false, bf.v, (short)0, dacc, false, false);
  }
#pragma unroll
  for (int r = 0; r < 8; ++r) part[(w * 32 + lane) * 8 + r] = dacc[r];
  __syncthreads();
  if (w < 4) {
#pragma unroll
    for (int r = 0; r < 8; ++r) {
      float s = part[(w * 32 + lane) * 8 + r] + part[((w + 4) * 32 + lane) * 8 + r];
      out[(size_t)(b * NC + c0 + lm) * NTOK + n0 + r + 8 * h] = s;
    }
  }
}

// ---------------- 1x1 conv: out[b,o,n] = sum_c W[o,c] in[b,c,n] (+bias | BN+ReLU) ----
__global__ void __launch_bounds__(256) conv1x1_kernel(
    const float* __restrict__ in, const float* __restrict__ W,
    const float* __restrict__ bias,
    const float* __restrict__ bnG, const float* __restrict__ bnBe,
    const float* __restrict__ bnM, const float* __restrict__ bnV,
    float* __restrict__ out, int Cin, int coutTotal, int coutBase)
{
  const int n = blockIdx.x * 256 + threadIdx.x;
  const int b = blockIdx.y;
  float acc[64];
#pragma unroll
  for (int o = 0; o < 64; ++o) acc[o] = 0.f;
  const float* inb = in + (size_t)b * Cin * NTOK + n;
  for (int c = 0; c < Cin; ++c) {
    const float xv = inb[(size_t)c * NTOK];
#pragma unroll
    for (int o = 0; o < 64; ++o) acc[o] += W[o * Cin + c] * xv;
  }
  float* ob = out + ((size_t)b * coutTotal + coutBase) * NTOK + n;
  if (bias) {
#pragma unroll
    for (int o = 0; o < 64; ++o) ob[(size_t)o * NTOK] = acc[o] + bias[o];
  } else {
#pragma unroll
    for (int o = 0; o < 64; ++o) {
      const float sc = bnG[o] / sqrtf(bnV[o] + 1e-5f);
      const float sh = bnBe[o] - bnM[o] * sc;
      ob[(size_t)o * NTOK] = fmaxf(acc[o] * sc + sh, 0.f);
    }
  }
}

// ---------------- 3x3 SAME conv + BN + ReLU ----------------
__global__ void __launch_bounds__(256) conv3x3_bn_relu_kernel(
    const float* __restrict__ in, const float* __restrict__ W,  // [64][64][3][3]
    const float* __restrict__ g, const float* __restrict__ be,
    const float* __restrict__ m, const float* __restrict__ v,
    float* __restrict__ out, int coutTotal, int coutBase)
{
  const int bo = blockIdx.x;
  const int b = bo >> 6, o = bo & 63;
  const int hw = blockIdx.y * 256 + threadIdx.x;
  const int hh = hw / HW48, ww = hw % HW48;
  float acc = 0.f;
  const float* inb = in + (size_t)b * 64 * NTOK;
  const float* wb = W + (size_t)o * 64 * 9;
  for (int c = 0; c < 64; ++c) {
    const float* ic = inb + c * NTOK;
    const float* wc = wb + c * 9;
#pragma unroll
    for (int kh = 0; kh < 3; ++kh) {
      const int h2 = hh + kh - 1;
      if (h2 < 0 || h2 >= HW48) continue;
#pragma unroll
      for (int kw = 0; kw < 3; ++kw) {
        const int w2 = ww + kw - 1;
        if (w2 < 0 || w2 >= HW48) continue;
        acc += wc[kh * 3 + kw] * ic[h2 * HW48 + w2];
      }
    }
  }
  const float sc = g[o] / sqrtf(v[o] + 1e-5f);
  const float sh = be[o] - m[o] * sc;
  out[((size_t)b * coutTotal + coutBase + o) * NTOK + hw] = fmaxf(acc * sc + sh, 0.f);
}

// ---------------- 3x3 avg pool, SAME, /9 ----------------
__global__ void __launch_bounds__(256) pool_kernel(const float* __restrict__ in,
                                                   float* __restrict__ out)
{
  const int bc = blockIdx.x;
  const int hw = blockIdx.y * 256 + threadIdx.x;
  const int hh = hw / HW48, ww = hw % HW48;
  const float* ib = in + (size_t)bc * NTOK;
  float s = 0.f;
#pragma unroll
  for (int dh = -1; dh <= 1; ++dh) {
    const int h2 = hh + dh;
    if (h2 < 0 || h2 >= HW48) continue;
#pragma unroll
    for (int dw = -1; dw <= 1; ++dw) {
      const int w2 = ww + dw;
      if (w2 < 0 || w2 >= HW48) continue;
      s += ib[h2 * HW48 + w2];
    }
  }
  out[(size_t)bc * NTOK + hw] = s * (1.0f / 9.0f);
}

// ---------------- f32 -> f16 converters ----------------
// transpose: in [B][64][N] -> out [B][N][64], with scale (for q: 1/sqrt(C))
__global__ void __launch_bounds__(256) cvtT_kernel(const float* __restrict__ in,
                                                   _Float16* __restrict__ out, float scale)
{
  const int idx = blockIdx.x * 256 + threadIdx.x;  // = ((b*N)+n)*64 + c
  const int c = idx & 63;
  const int n = (idx >> 6) % NTOK;
  const int b = idx / (NTOK * 64);
  out[idx] = (_Float16)(in[((size_t)b * 64 + c) * NTOK + n] * scale);
}

__global__ void __launch_bounds__(256) cvt_kernel(const float* __restrict__ in,
                                                  _Float16* __restrict__ out)
{
  const int idx = blockIdx.x * 256 + threadIdx.x;
  out[idx] = (_Float16)in[idx];
}

// ---------------- host orchestration ----------------
extern "C" void kernel_launch(void* const* d_in, const int* in_sizes, int n_in,
                              void* d_out, int out_size, void* d_ws, size_t ws_size,
                              hipStream_t stream) {
  (void)in_sizes; (void)n_in; (void)out_size; (void)ws_size;

  const float* x = (const float*)d_in[0];
  const float* y = (const float*)d_in[1];
  // 1x1 qkv / projection params (w,b pairs), in params insertion order
  const float* W_qx = (const float*)d_in[2];  const float* b_qx = (const float*)d_in[3];
  const float* W_kx = (const float*)d_in[4];  const float* b_kx = (const float*)d_in[5];
  const float* W_vx = (const float*)d_in[6];  const float* b_vx = (const float*)d_in[7];
  const float* W_xx = (const float*)d_in[8];  const float* b_xx = (const float*)d_in[9];
  const float* W_xy = (const float*)d_in[10]; const float* b_xy = (const float*)d_in[11];
  const float* W_xxp = (const float*)d_in[12]; const float* b_xxp = (const float*)d_in[13];
  const float* W_xyp = (const float*)d_in[14]; const float* b_xyp = (const float*)d_in[15];
  const float* W_qy = (const float*)d_in[16]; const float* b_qy = (const float*)d_in[17];
  const float* W_ky = (const float*)d_in[18]; const float* b_ky = (const float*)d_in[19];
  const float* W_vy = (const float*)d_in[20]; const float* b_vy = (const float*)d_in[21];
  const float* W_yy = (const float*)d_in[22]; const float* b_yy = (const float*)d_in[23];
  const float* W_yx = (const float*)d_in[24]; const float* b_yx = (const float*)d_in[25];
  const float* W_yyp = (const float*)d_in[26]; const float* b_yyp = (const float*)d_in[27];
  const float* W_yxp = (const float*)d_in[28]; const float* b_yxp = (const float*)d_in[29];
  // cat blocks
  const float* cat_w1 = (const float*)d_in[30];
  const float* cat_g1 = (const float*)d_in[31]; const float* cat_be1 = (const float*)d_in[32];
  const float* cat_m1 = (const float*)d_in[33]; const float* cat_v1 = (const float*)d_in[34];
  const float* cat_w2 = (const float*)d_in[35];
  const float* cat_g2 = (const float*)d_in[36]; const float* cat_be2 = (const float*)d_in[37];
  const float* cat_m2 = (const float*)d_in[38]; const float* cat_v2 = (const float*)d_in[39];
  const float* catp_w1 = (const float*)d_in[40];
  const float* catp_g1 = (const float*)d_in[41]; const float* catp_be1 = (const float*)d_in[42];
  const float* catp_m1 = (const float*)d_in[43]; const float* catp_v1 = (const float*)d_in[44];
  const float* catp_w2 = (const float*)d_in[45];
  const float* catp_g2 = (const float*)d_in[46]; const float* catp_be2 = (const float*)d_in[47];
  const float* catp_m2 = (const float*)d_in[48]; const float* catp_v2 = (const float*)d_in[49];
  const float* catf_w1 = (const float*)d_in[50];
  const float* catf_g1 = (const float*)d_in[51]; const float* catf_be1 = (const float*)d_in[52];
  const float* catf_m1 = (const float*)d_in[53]; const float* catf_v1 = (const float*)d_in[54];
  const float* catf_w2 = (const float*)d_in[55];
  const float* catf_g2 = (const float*)d_in[56]; const float* catf_be2 = (const float*)d_in[57];
  const float* catf_m2 = (const float*)d_in[58]; const float* catf_v2 = (const float*)d_in[59];
  const int* topkP = (const int*)d_in[60];

  // ---- workspace carve ----
  char* ws = (char*)d_ws;
  size_t off = 0;
  auto carve = [&](size_t bytes) -> void* {
    void* p = (void*)(ws + off);
    off += (bytes + 255) & ~(size_t)255;
    return p;
  };
  const size_t BCN4 = (size_t)NB * NC * NTOK * 4;   // 2.36 MB
  const size_t BCN2 = (size_t)NB * NC * NTOK * 2;
  float* qF[2]; float* kF[2]; float* vF[2];
  _Float16* qTh[2]; _Float16* kTh[2]; _Float16* vHh[2];
  for (int i = 0; i < 2; ++i) {
    qF[i] = (float*)carve(BCN4); kF[i] = (float*)carve(BCN4); vF[i] = (float*)carve(BCN4);
    qTh[i] = (_Float16*)carve(BCN2); kTh[i] = (_Float16*)carve(BCN2); vHh[i] = (_Float16*)carve(BCN2);
  }
  float* attO    = (float*)carve(BCN4);
  float* catin   = (float*)carve((size_t)NB * 256 * NTOK * 4);
  float* t1      = (float*)carve(BCN4);
  float* finalIn = (float*)carve((size_t)NB * 128 * NTOK * 4);
  float* px      = (float*)carve(BCN4);
  float* py      = (float*)carve(BCN4);

  // allow >64KB dynamic LDS for the attention kernel (MI455X: 320KB/WGP)
  (void)hipFuncSetAttribute(reinterpret_cast<const void*>(attend_kernel),
                            hipFuncAttributeMaxDynamicSharedMemorySize, SM_TOTAL);

  const dim3 g1(NTOK / 256, NB);       // (9, 4) for per-position kernels
  const dim3 ga(NTOK / 16, NB);        // (144, 4) attention tiles
  const dim3 gc(NB * NC, NTOK / 256);  // (256, 9) per (b,o) spatial kernels
  const int cvtBlocks = (NB * NC * NTOK) / 256;

  auto branch = [&](const float* X, const float* Y,
                    const float* Wxx_, const float* bxx_,
                    const float* Wxy_, const float* bxy_,
                    const float* Wyy_, const float* byy_,
                    const float* Wyx_, const float* byx_) {
    // q,k,v for both streams (shared weights across plain/pooled branches)
    conv1x1_kernel<<<g1, 256, 0, stream>>>(X, W_qx, b_qx, nullptr, nullptr, nullptr, nullptr, qF[0], 64, 64, 0);
    conv1x1_kernel<<<g1, 256, 0, stream>>>(X, W_kx, b_kx, nullptr, nullptr, nullptr, nullptr, kF[0], 64, 64, 0);
    conv1x1_kernel<<<g1, 256, 0, stream>>>(X, W_vx, b_vx, nullptr, nullptr, nullptr, nullptr, vF[0], 64, 64, 0);
    conv1x1_kernel<<<g1, 256, 0, stream>>>(Y, W_qy, b_qy, nullptr, nullptr, nullptr, nullptr, qF[1], 64, 64, 0);
    conv1x1_kernel<<<g1, 256, 0, stream>>>(Y, W_ky, b_ky, nullptr, nullptr, nullptr, nullptr, kF[1], 64, 64, 0);
    conv1x1_kernel<<<g1, 256, 0, stream>>>(Y, W_vy, b_vy, nullptr, nullptr, nullptr, nullptr, vF[1], 64, 64, 0);
    // f16 conversions (q pre-scaled by 1/sqrt(C)=0.125; q,k transposed to [B][N][C])
    cvtT_kernel<<<cvtBlocks, 256, 0, stream>>>(qF[0], qTh[0], 0.125f);
    cvtT_kernel<<<cvtBlocks, 256, 0, stream>>>(kF[0], kTh[0], 1.0f);
    cvt_kernel <<<cvtBlocks, 256, 0, stream>>>(vF[0], vHh[0]);
    cvtT_kernel<<<cvtBlocks, 256, 0, stream>>>(qF[1], qTh[1], 0.125f);
    cvtT_kernel<<<cvtBlocks, 256, 0, stream>>>(kF[1], kTh[1], 1.0f);
    cvt_kernel <<<cvtBlocks, 256, 0, stream>>>(vF[1], vHh[1]);
    // 4 attentions, each projected (1x1+bias) into the concat buffer channel slots
    attend_kernel<<<ga, 256, SM_TOTAL, stream>>>(qTh[0], kTh[0], vHh[0], attO, topkP);
    conv1x1_kernel<<<g1, 256, 0, stream>>>(attO, Wxx_, bxx_, nullptr, nullptr, nullptr, nullptr, catin, 64, 256, 0);
    attend_kernel<<<ga, 256, SM_TOTAL, stream>>>(qTh[0], kTh[1], vHh[1], attO, topkP);
    conv1x1_kernel<<<g1, 256, 0, stream>>>(attO, Wxy_, bxy_, nullptr, nullptr, nullptr, nullptr, catin, 64, 256, 64);
    attend_kernel<<<ga, 256, SM_TOTAL, stream>>>(qTh[1], kTh[1], vHh[1], attO, topkP);
    conv1x1_kernel<<<g1, 256, 0, stream>>>(attO, Wyy_, byy_, nullptr, nullptr, nullptr, nullptr, catin, 64, 256, 128);
    attend_kernel<<<ga, 256, SM_TOTAL, stream>>>(qTh[1], kTh[0], vHh[0], attO, topkP);
    conv1x1_kernel<<<g1, 256, 0, stream>>>(attO, Wyx_, byx_, nullptr, nullptr, nullptr, nullptr, catin, 64, 256, 192);
  };

  // ---- main branch ----
  branch(x, y, W_xx, b_xx, W_xy, b_xy, W_yy, b_yy, W_yx, b_yx);
  conv1x1_kernel<<<g1, 256, 0, stream>>>(catin, cat_w1, nullptr, cat_g1, cat_be1, cat_m1, cat_v1, t1, 256, 64, 0);
  conv3x3_bn_relu_kernel<<<gc, 256, 0, stream>>>(t1, cat_w2, cat_g2, cat_be2, cat_m2, cat_v2, finalIn, 128, 0);

  // ---- pooled branch ----
  pool_kernel<<<gc, 256, 0, stream>>>(x, px);
  pool_kernel<<<gc, 256, 0, stream>>>(y, py);
  branch(px, py, W_xxp, b_xxp, W_xyp, b_xyp, W_yyp, b_yyp, W_yxp, b_yxp);
  conv1x1_kernel<<<g1, 256, 0, stream>>>(catin, catp_w1, nullptr, catp_g1, catp_be1, catp_m1, catp_v1, t1, 256, 64, 0);
  conv3x3_bn_relu_kernel<<<gc, 256, 0, stream>>>(t1, catp_w2, catp_g2, catp_be2, catp_m2, catp_v2, finalIn, 128, 64);

  // ---- final cat block -> d_out ----
  conv1x1_kernel<<<g1, 256, 0, stream>>>(finalIn, catf_w1, nullptr, catf_g1, catf_be1, catf_m1, catf_v1, t1, 128, 64, 0);
  conv3x3_bn_relu_kernel<<<gc, 256, 0, stream>>>(t1, catf_w2, catf_g2, catf_be2, catf_m2, catf_v2, (float*)d_out, 64, 0);
}
